// SimpleSwin_8126078124483
// MI455X (gfx1250) — compile-verified, matrix operand order checked by reading, not run on hardware
//
#include <hip/hip_runtime.h>

// ---------------------------------------------------------------------------
// Fused Swin forward for MI455X (gfx1250, wave32, WMMA), compile-time C.
// GEMMs: v_wmma_f32_16x16x32_f16, f32 accumulate.
// A fragments: 2x ds_load_b128 from LDS. B fragments: 2x global_load_b128
// from weights repacked once per launch into fragment order (f16).
// ---------------------------------------------------------------------------

typedef __attribute__((ext_vector_type(16))) _Float16 v16h;
typedef __attribute__((ext_vector_type(8)))  _Float16 v8h;
typedef __attribute__((ext_vector_type(8)))  float    v8f;

#define NHEADS 6
#define LN_EPS 1e-5f

__device__ __forceinline__ v8f wmma_f32_f16(v16h a, v16h b, v8f c) {
  return __builtin_amdgcn_wmma_f32_16x16x32_f16(false, a, false, b, (short)0, c,
                                                false, false);
}

// ---------------------------------------------------------------------------
// Weight repack: W (K x N, f32, row-major) -> fragment-ordered f16 tiles.
// Tile (kt, nt) holds a 32x16 panel; lane l's 16 B-fragment slots are the 16
// contiguous halves at tile*512 + l*16.  Slot j of lane (n_loc, half) maps to
// k = kt*32 + (j>>3)*16 + (j&7) + half*8  (ISA 7.12.2 16-bit operand layout).
// ---------------------------------------------------------------------------
__global__ __launch_bounds__(128) void repack_w(const float* __restrict__ W,
                                                _Float16* __restrict__ Wp,
                                                int K, int N) {
  const int NT = N >> 4;
  const int kt = blockIdx.x / NT, nt = blockIdx.x % NT;
#pragma unroll
  for (int q = 0; q < 4; ++q) {
    const int idx  = threadIdx.x * 4 + q;        // 0..511
    const int lane = idx >> 4, slot = idx & 15;
    const int half = lane >> 4;
    const int n    = nt * 16 + (lane & 15);
    const int k    = kt * 32 + ((slot >> 3) << 4) + (slot & 7) + (half << 3);
    const float v  = (k < K) ? W[(size_t)k * N + n] : 0.0f;
    Wp[(size_t)blockIdx.x * 512 + idx] = (_Float16)v;
  }
}

// 16x16 output tile of A(16xK, f16 LDS, row stride K) @ Wp (repacked), col nt.
template <int K>
__device__ __forceinline__ v8f gemm_tile(const _Float16* __restrict__ A,
                                         const _Float16* __restrict__ Wp,
                                         int NT, int nt) {
  const int lane = threadIdx.x & 31;
  const int half = lane >> 4;
  const int l15  = lane & 15;
  constexpr int KT = (K + 31) / 32;
  v8f acc = {};
#pragma unroll
  for (int kt = 0; kt < KT; ++kt) {
    const int k0 = kt * 32;
    const _Float16* ar = &A[l15 * K + k0 + half * 8];
    v8h alo = *(const v8h*)ar;                      // slots 0..7
    v8h ahi = {};
    if (k0 + 32 <= K) ahi = *(const v8h*)(ar + 16); // slots 8..15 (or K-pad)
    const _Float16* bp = &Wp[((size_t)(kt * NT + nt)) * 512 + lane * 16];
    v8h blo = *(const v8h*)bp;
    v8h bhi = *(const v8h*)(bp + 8);
    if (kt + 1 < KT)
      __builtin_prefetch(&Wp[((size_t)((kt + 1) * NT + nt)) * 512 + lane * 16], 0, 1);
    v16h a, b;
#pragma unroll
    for (int j = 0; j < 8; ++j) {
      a[j] = alo[j]; a[j + 8] = ahi[j];
      b[j] = blo[j]; b[j + 8] = bhi[j];
    }
    acc = wmma_f32_f16(a, b, acc);
  }
  return acc;
}

// ---------------------------------------------------------------------------
// Fused window attention: LN -> qkv -> softmax(QK^T)V -> proj -> +residual.
// One workgroup (4 waves) per 4x4 window (16 tokens).
// ---------------------------------------------------------------------------
template <int C>
__global__ __launch_bounds__(128) void swin_attn_kernel(
    float* __restrict__ x, int B, int H, int W,
    const float* __restrict__ n1_g, const float* __restrict__ n1_b,
    const _Float16* __restrict__ qkv_wp, const float* __restrict__ qkv_b,
    const _Float16* __restrict__ proj_wp, const float* __restrict__ proj_b) {
  constexpr int d   = C / NHEADS;
  constexpr int ld3 = 3 * C;
  constexpr int NTQ = ld3 / 16;
  constexpr int NTP = C / 16;
  constexpr int CV  = C / 4;

  __shared__ alignas(16) float    xres[16 * C];
  __shared__ alignas(16) _Float16 hln[16 * C];
  __shared__ alignas(16) _Float16 qkvs[16 * ld3 + 64];
  __shared__ alignas(16) _Float16 aout[16 * C];
  __shared__ alignas(16) _Float16 Pl[4][256];
  __shared__ float mu[16], rs[16];

  const int tid  = threadIdx.x;
  const int wave = tid >> 5;
  const int lane = tid & 31;
  const int half = lane >> 4;
  const int l15  = lane & 15;

  const int Wq = W >> 2, Hq = H >> 2;
  const int b   = blockIdx.x / (Hq * Wq);
  const int rem = blockIdx.x % (Hq * Wq);
  const int wh  = rem / Wq, ww = rem % Wq;

  // ---- cooperative vectorized gather of the 16 window tokens ----
  for (int i = tid; i < 16 * CV; i += 128) {
    const int m = i / CV, c4 = i % CV;
    const int y = wh * 4 + (m >> 2), xx = ww * 4 + (m & 3);
    *(float4*)(xres + m * C + c4 * 4) =
        *(const float4*)(x + ((size_t)(b * H + y) * W + xx) * C + c4 * 4);
  }
  __syncthreads();
  if (tid < 16) {
    float s = 0.f, s2 = 0.f;
    for (int c = 0; c < C; ++c) {
      const float v = xres[tid * C + c];
      s += v; s2 += v * v;
    }
    const float m = s / C;
    mu[tid] = m;
    rs[tid] = rsqrtf(s2 / C - m * m + LN_EPS);
  }
  __syncthreads();
  for (int i = tid; i < 16 * C; i += 128) {
    const int m = i / C, c = i % C;
    hln[i] = (_Float16)((xres[i] - mu[m]) * rs[m] * n1_g[c] + n1_b[c]);
  }
  __syncthreads();

  // ---- qkv GEMM: 16xC @ Cx3C ----
  for (int nt = wave; nt < NTQ; nt += 4) {
    v8f acc = gemm_tile<C>(hln, qkv_wp, NTQ, nt);
    const int n   = nt * 16 + l15;
    const float g = qkv_b[n];
    const int mb  = half << 3;
#pragma unroll
    for (int r = 0; r < 8; ++r)
      qkvs[(mb + r) * ld3 + n] = (_Float16)(acc[r] + g);
  }
  __syncthreads();

  // ---- per-head attention (one wave per head) ----
  _Float16* P = Pl[wave];
  const float scale = rsqrtf((float)d);
  for (int head = wave; head < NHEADS; head += 4) {
    const int qo = head * d, ko = C + head * d, vo = 2 * C + head * d;
    // S = q @ k^T ; q rows and k^T columns are contiguous f16 runs in LDS.
    const _Float16* qr = &qkvs[l15 * ld3 + qo + half * 8];
    const _Float16* kr = &qkvs[l15 * ld3 + ko + half * 8];
    v8h qlo = *(const v8h*)qr;
    v8h klo = *(const v8h*)kr;
    if (d == 8 && half) {            // stage1: upper half of slots is K-pad
#pragma unroll
      for (int j = 0; j < 8; ++j) { qlo[j] = (_Float16)0; klo[j] = (_Float16)0; }
    }
    v8h qhi = {}, khi = {};
    if (d == 32) {                   // stage3: slots 8..15 hold K=16..31
      qhi = *(const v8h*)(qr + 16);
      khi = *(const v8h*)(kr + 16);
    }
    v16h qa, kb;
#pragma unroll
    for (int j = 0; j < 8; ++j) {
      qa[j] = qlo[j]; qa[j + 8] = qhi[j];
      kb[j] = klo[j]; kb[j + 8] = khi[j];
    }
    v8f sc = {};
    sc = wmma_f32_f16(qa, kb, sc);
    // row-wise softmax: row m = r + 8*half lives in one 16-lane half
#pragma unroll
    for (int r = 0; r < 8; ++r) {
      float v = sc[r] * scale;
      float mx = v;
      mx = fmaxf(mx, __shfl_xor(mx, 1));
      mx = fmaxf(mx, __shfl_xor(mx, 2));
      mx = fmaxf(mx, __shfl_xor(mx, 4));
      mx = fmaxf(mx, __shfl_xor(mx, 8));
      float e = __expf(v - mx);
      float su = e;
      su += __shfl_xor(su, 1);
      su += __shfl_xor(su, 2);
      su += __shfl_xor(su, 4);
      su += __shfl_xor(su, 8);
      P[((half << 3) + r) * 16 + l15] = (_Float16)(e / su);
    }
    // out = P @ v  (16 keys pad to K=32; d columns in 16-wide tiles)
    constexpr int NTV = (d + 15) / 16;
#pragma unroll
    for (int nt = 0; nt < NTV; ++nt) {
      const v8h plo = *(const v8h*)&P[l15 * 16 + half * 8];
      v16h pa, vb;
      const int vc = nt * 16 + l15;
#pragma unroll
      for (int j = 0; j < 8; ++j) {
        pa[j] = plo[j]; pa[j + 8] = (_Float16)0;
        const _Float16 t = qkvs[(j + half * 8) * ld3 + vo + vc];
        vb[j] = (vc < d) ? t : (_Float16)0;   // folds away unless d==8
        vb[j + 8] = (_Float16)0;
      }
      v8f o = {};
      o = wmma_f32_f16(pa, vb, o);
      if (vc < d) {
#pragma unroll
        for (int r = 0; r < 8; ++r)
          aout[((half << 3) + r) * C + head * d + vc] = (_Float16)o[r];
      }
    }
  }
  __syncthreads();

  // ---- proj GEMM + bias + residual -> global ----
  for (int nt = wave; nt < NTP; nt += 4) {
    v8f acc = gemm_tile<C>(aout, proj_wp, NTP, nt);
    const int n   = nt * 16 + l15;
    const float g = proj_b[n];
    const int mb  = half << 3;
#pragma unroll
    for (int r = 0; r < 8; ++r) {
      const int m = mb + r;
      const int y = wh * 4 + (m >> 2), xx = ww * 4 + (m & 3);
      x[((size_t)(b * H + y) * W + xx) * C + n] = xres[m * C + n] + acc[r] + g;
    }
  }
}

// ---------------------------------------------------------------------------
// Fused MLP: LN -> fc1 -> GELU(exact) -> fc2 -> +residual, per 16-row tile.
// ---------------------------------------------------------------------------
template <int C>
__global__ __launch_bounds__(128) void swin_mlp_kernel(
    float* __restrict__ x,
    const float* __restrict__ n2_g, const float* __restrict__ n2_b,
    const _Float16* __restrict__ fc1_wp, const float* __restrict__ fc1_b,
    const _Float16* __restrict__ fc2_wp, const float* __restrict__ fc2_b) {
  constexpr int C4  = 4 * C;
  constexpr int NT1 = C4 / 16;
  constexpr int NT2 = C / 16;

  __shared__ alignas(16) float    xres[16 * C];
  __shared__ alignas(16) _Float16 hln[16 * C];
  __shared__ alignas(16) _Float16 mid[16 * C4];
  __shared__ float mu[16], rs[16];

  const int tid  = threadIdx.x;
  const int wave = tid >> 5;
  const int lane = tid & 31;
  const int half = lane >> 4;
  const int l15  = lane & 15;
  const size_t row0 = (size_t)blockIdx.x * 16;

  for (int i = tid; i < 4 * C; i += 128)   // 16*C floats = 4*C float4s
    *(float4*)(xres + i * 4) = *(const float4*)(x + row0 * C + i * 4);
  __syncthreads();
  if (tid < 16) {
    float s = 0.f, s2 = 0.f;
    for (int c = 0; c < C; ++c) {
      const float v = xres[tid * C + c];
      s += v; s2 += v * v;
    }
    const float m = s / C;
    mu[tid] = m;
    rs[tid] = rsqrtf(s2 / C - m * m + LN_EPS);
  }
  __syncthreads();
  for (int i = tid; i < 16 * C; i += 128) {
    const int m = i / C, c = i % C;
    hln[i] = (_Float16)((xres[i] - mu[m]) * rs[m] * n2_g[c] + n2_b[c]);
  }
  __syncthreads();

  for (int nt = wave; nt < NT1; nt += 4) {
    v8f acc = gemm_tile<C>(hln, fc1_wp, NT1, nt);
    const int n   = nt * 16 + l15;
    const float g = fc1_b[n];
    const int mb  = half << 3;
#pragma unroll
    for (int r = 0; r < 8; ++r) {
      const float v = acc[r] + g;
      mid[(mb + r) * C4 + n] = (_Float16)(0.5f * v * (1.0f + erff(v * 0.70710678f)));
    }
  }
  __syncthreads();

  for (int nt = wave; nt < NT2; nt += 4) {
    v8f acc = gemm_tile<C4>(mid, fc2_wp, NT2, nt);
    const int n   = nt * 16 + l15;
    const float g = fc2_b[n];
    const int mb  = half << 3;
#pragma unroll
    for (int r = 0; r < 8; ++r)
      x[(row0 + mb + r) * C + n] = xres[(mb + r) * C + n] + acc[r] + g;
  }
}

// ---------------------------------------------------------------------------
// Patch merge: gather 2x2 (4C) -> LN -> @ red_w (4C x 2C).
// ---------------------------------------------------------------------------
template <int C>
__global__ __launch_bounds__(128) void patch_merge_kernel(
    const float* __restrict__ xin, float* __restrict__ xout, int B, int H, int W,
    const float* __restrict__ n_g, const float* __restrict__ n_b,
    const _Float16* __restrict__ red_wp) {
  constexpr int C4 = 4 * C, Co = 2 * C;
  constexpr int NTO = Co / 16;
  constexpr int CV  = C / 4;

  __shared__ alignas(16) float    mg[16 * C4];
  __shared__ alignas(16) _Float16 hln[16 * C4];
  __shared__ float mu[16], rs[16];

  const int tid  = threadIdx.x;
  const int wave = tid >> 5;
  const int lane = tid & 31;
  const int half = lane >> 4;
  const int l15  = lane & 15;
  const int Ho = H >> 1, Wo = W >> 1;
  const size_t t0 = (size_t)blockIdx.x * 16;

  for (int i = tid; i < 16 * 4 * CV; i += 128) {
    const int m = i / (4 * CV), rem2 = i % (4 * CV);
    const int part = rem2 / CV, c4 = rem2 % CV;
    const int dy = part >> 1, dx = part & 1;          // [x0,x1,x2,x3]
    const size_t tok = t0 + m;
    const int bb = (int)(tok / ((size_t)Ho * Wo));
    const int r2 = (int)(tok % ((size_t)Ho * Wo));
    const int oy = r2 / Wo, ox = r2 % Wo;
    *(float4*)(mg + m * C4 + part * C + c4 * 4) = *(const float4*)(
        xin + ((size_t)(bb * H + 2 * oy + dy) * W + (2 * ox + dx)) * C + c4 * 4);
  }
  __syncthreads();
  if (tid < 16) {
    float s = 0.f, s2 = 0.f;
    for (int c = 0; c < C4; ++c) {
      const float v = mg[tid * C4 + c];
      s += v; s2 += v * v;
    }
    const float m = s / C4;
    mu[tid] = m;
    rs[tid] = rsqrtf(s2 / C4 - m * m + LN_EPS);
  }
  __syncthreads();
  for (int i = tid; i < 16 * C4; i += 128) {
    const int m = i / C4, c = i % C4;
    hln[i] = (_Float16)((mg[i] - mu[m]) * rs[m] * n_g[c] + n_b[c]);
  }
  __syncthreads();

  for (int nt = wave; nt < NTO; nt += 4) {
    v8f acc = gemm_tile<C4>(hln, red_wp, NTO, nt);
    const int n  = nt * 16 + l15;
    const int mb = half << 3;
#pragma unroll
    for (int r = 0; r < 8; ++r) xout[(t0 + mb + r) * Co + n] = acc[r];
  }
}

// ---------------------------------------------------------------------------
// Final LN -> mean pool (16x16 tokens) -> head (192 -> 10).
// ---------------------------------------------------------------------------
__global__ __launch_bounds__(256) void head_kernel(
    const float* __restrict__ x, float* __restrict__ out,
    const float* __restrict__ ng, const float* __restrict__ nb,
    const float* __restrict__ hw, const float* __restrict__ hb) {
  __shared__ float pooled[192];
  const int b = blockIdx.x, t = threadIdx.x;
  if (t < 192) pooled[t] = 0.f;
  __syncthreads();
  const float* src = x + ((size_t)b * 256 + t) * 192;
  float s = 0.f, s2 = 0.f;
  for (int c = 0; c < 192; ++c) {
    const float v = src[c];
    s += v; s2 += v * v;
  }
  const float m = s * (1.f / 192.f);
  const float r = rsqrtf(s2 * (1.f / 192.f) - m * m + LN_EPS);
  for (int c = 0; c < 192; ++c)
    atomicAdd(&pooled[c], ((src[c] - m) * r * ng[c] + nb[c]) * (1.f / 256.f));
  __syncthreads();
  if (t < 10) {
    float acc = hb[t];
    for (int c = 0; c < 192; ++c) acc += pooled[c] * hw[c * 10 + t];
    out[b * 10 + t] = acc;
  }
}

// ---------------------------------------------------------------------------
// Host side
// ---------------------------------------------------------------------------
struct BlkF {
  const float *n1_g, *n1_b, *qkv_w, *qkv_b, *proj_w, *proj_b;
  const float *n2_g, *n2_b, *fc1_w, *fc1_b, *fc2_w, *fc2_b;
};
struct BlkP {
  const float *n1_g, *n1_b, *qkv_b, *proj_b, *n2_g, *n2_b, *fc1_b, *fc2_b;
  const _Float16 *qkv_wp, *proj_wp, *fc1_wp, *fc2_wp;
};

template <int C>
static void run_block(float* xb, int B, int H, int W, const BlkP& p,
                      hipStream_t stream) {
  const int nW = B * (H / 4) * (W / 4);
  swin_attn_kernel<C><<<nW, 128, 0, stream>>>(xb, B, H, W, p.n1_g, p.n1_b,
                                              p.qkv_wp, p.qkv_b, p.proj_wp,
                                              p.proj_b);
  const int Ntok = B * H * W;
  swin_mlp_kernel<C><<<Ntok / 16, 128, 0, stream>>>(xb, p.n2_g, p.n2_b,
                                                    p.fc1_wp, p.fc1_b,
                                                    p.fc2_wp, p.fc2_b);
}

extern "C" void kernel_launch(void* const* d_in, const int* in_sizes, int n_in,
                              void* d_out, int out_size, void* d_ws,
                              size_t ws_size, hipStream_t stream) {
  (void)in_sizes; (void)n_in; (void)out_size; (void)ws_size;
  int idx = 0;
  auto nextp = [&]() -> const float* { return (const float*)d_in[idx++]; };

  // Inputs in setup_inputs() insertion order, recursing in order.
  const float* x_in = nextp();
  auto readBlk = [&]() {
    BlkF p;
    p.n1_g = nextp(); p.n1_b = nextp();
    p.qkv_w = nextp(); p.qkv_b = nextp();
    p.proj_w = nextp(); p.proj_b = nextp();
    p.n2_g = nextp(); p.n2_b = nextp();
    p.fc1_w = nextp(); p.fc1_b = nextp();
    p.fc2_w = nextp(); p.fc2_b = nextp();
    return p;
  };
  BlkF s1[2]; s1[0] = readBlk(); s1[1] = readBlk();
  const float *m1_ng = nextp(), *m1_nb = nextp(), *m1_rw = nextp();
  BlkF s2[2]; s2[0] = readBlk(); s2[1] = readBlk();
  const float *m2_ng = nextp(), *m2_nb = nextp(), *m2_rw = nextp();
  BlkF s3[4]; for (int i = 0; i < 4; ++i) s3[i] = readBlk();
  const float *norm_g = nextp(), *norm_b = nextp();
  const float *head_w = nextp(), *head_b = nextp();

  // Workspace bump allocator: activations + repacked f16 weights.
  size_t off = 0;
  auto bump = [&](size_t bytes) -> void* {
    void* p = (char*)d_ws + off;
    off = (off + bytes + 255) & ~(size_t)255;
    return p;
  };
  const int B = 64;
  float* buf0 = (float*)bump((size_t)B * 64 * 64 * 48 * sizeof(float));
  float* buf1 = (float*)bump((size_t)B * 32 * 32 * 96 * sizeof(float));

  auto repack = [&](const float* W, int K, int N) -> _Float16* {
    const int KT = (K + 31) / 32, NT = N / 16;
    _Float16* dst = (_Float16*)bump((size_t)KT * NT * 512 * sizeof(_Float16));
    repack_w<<<KT * NT, 128, 0, stream>>>(W, dst, K, N);
    return dst;
  };
  auto packBlk = [&](const BlkF& f, int C) {
    BlkP p;
    p.n1_g = f.n1_g; p.n1_b = f.n1_b; p.qkv_b = f.qkv_b; p.proj_b = f.proj_b;
    p.n2_g = f.n2_g; p.n2_b = f.n2_b; p.fc1_b = f.fc1_b; p.fc2_b = f.fc2_b;
    p.qkv_wp  = repack(f.qkv_w, C, 3 * C);
    p.proj_wp = repack(f.proj_w, C, C);
    p.fc1_wp  = repack(f.fc1_w, C, 4 * C);
    p.fc2_wp  = repack(f.fc2_w, 4 * C, C);
    return p;
  };

  BlkP p1[2] = {packBlk(s1[0], 48), packBlk(s1[1], 48)};
  const _Float16* m1_wp = repack(m1_rw, 192, 96);
  BlkP p2[2] = {packBlk(s2[0], 96), packBlk(s2[1], 96)};
  const _Float16* m2_wp = repack(m2_rw, 384, 192);
  BlkP p3[4] = {packBlk(s3[0], 192), packBlk(s3[1], 192), packBlk(s3[2], 192),
                packBlk(s3[3], 192)};

  hipMemcpyAsync(buf0, x_in, (size_t)B * 64 * 64 * 48 * sizeof(float),
                 hipMemcpyDeviceToDevice, stream);

  run_block<48>(buf0, B, 64, 64, p1[0], stream);
  run_block<48>(buf0, B, 64, 64, p1[1], stream);
  patch_merge_kernel<48><<<(B * 32 * 32) / 16, 128, 0, stream>>>(
      buf0, buf1, B, 64, 64, m1_ng, m1_nb, m1_wp);
  run_block<96>(buf1, B, 32, 32, p2[0], stream);
  run_block<96>(buf1, B, 32, 32, p2[1], stream);
  patch_merge_kernel<96><<<(B * 16 * 16) / 16, 128, 0, stream>>>(
      buf1, buf0, B, 32, 32, m2_ng, m2_nb, m2_wp);
  for (int i = 0; i < 4; ++i) run_block<192>(buf0, B, 16, 16, p3[i], stream);
  head_kernel<<<B, 256, 0, stream>>>(buf0, (float*)d_out, norm_g, norm_b,
                                     head_w, head_b);
}
